// MoDLayer_27711128994006
// MI455X (gfx1250) — compile-verified, matrix operand order checked by reading, not run on hardware
//
#include <hip/hip_runtime.h>
#include <hip/hip_bf16.h>
#include <math.h>

// ---------------------------------------------------------------------------
// MoD Qwen2 block for MI455X (gfx1250, wave32, WMMA 16x16x32 f16).
// GEMMs: TDM (tensor_load_to_lds) double-buffered LDS tiles -> WMMA.
// Attention: flash-style streaming softmax, WMMA for QK^T and P*V.
// ---------------------------------------------------------------------------

typedef __attribute__((ext_vector_type(16))) _Float16 v16h;
typedef __attribute__((ext_vector_type(8)))  _Float16 v8h;
typedef __attribute__((ext_vector_type(8)))  float    v8f;
typedef __attribute__((ext_vector_type(4)))  unsigned int u32x4;
typedef __attribute__((ext_vector_type(8)))  int          i32x8;
typedef __attribute__((ext_vector_type(4)))  int          i32x4;

#ifndef __has_builtin
#define __has_builtin(x) 0
#endif
#if __has_builtin(__builtin_amdgcn_tensor_load_to_lds) && \
    __has_builtin(__builtin_amdgcn_s_wait_tensorcnt)
#define TDM_AVAILABLE 1
#else
#define TDM_AVAILABLE 0
#endif

#define B_   2
#define S_   2048
#define D_   2048
#define HQ_  16
#define HKV_ 8
#define HD_  128
#define FF_  8192
#define NTOK (B_ * S_)
#define KCAP 256   // GAMMA * S

// ---------------------------------------------------------------------------
// WMMA fragment helpers (CDNA5 ISA 7.12.2 layouts)
// 16-bit A 16x32: lane<16 -> row=lane, K = {k0..k0+7, k0+16..k0+23}
//                 lane>=16 -> row=lane-16, K = {k0+8..k0+15, k0+24..k0+31}
// B 32x16 mirrors A with row->column. C/D: VGPR r -> row r + 8*(lane>=16),
// lane&15 -> column.
// ---------------------------------------------------------------------------
__device__ __forceinline__ v16h frag_ld(const _Float16* __restrict__ base,
                                        int ld, int row0, int k0) {
  const int lane = threadIdx.x & 31;
  const _Float16* p = base + (size_t)(row0 + (lane & 15)) * (size_t)ld
                           + k0 + ((lane & 16) ? 8 : 0);
  v8h lo = *(const v8h*)(p);
  v8h hi = *(const v8h*)(p + 16);
  v16h r;
#pragma unroll
  for (int i = 0; i < 8; ++i) { r[i] = lo[i]; r[i + 8] = hi[i]; }
  return r;
}

__device__ __forceinline__ v8f wmma16(v16h a, v16h b, v8f c) {
  return __builtin_amdgcn_wmma_f32_16x16x32_f16(false, a, false, b,
                                                (short)0, c, false, false);
}

// ---------------------------------------------------------------------------
// TDM: DMA a 2D tile (32 elems x `rows` rows, f16) from global into LDS.
// Descriptor built per CDNA5 ISA section 8.3/8.4 (D# group0/group1 bitfields):
//   g0: count=1 | lds_addr | global_addr[56:0] | type=2
//   g1: data_size=1(2B) | tensor_dim0=K | tensor_dim1=rows | tile 32 x rows |
//       tensor_dim0_stride=K
// Issued once per wave (EXEC ignored for TDM); completion via TENSORcnt.
// ---------------------------------------------------------------------------
__device__ __forceinline__ void tdm_load_tile(unsigned lds_off,
                                              const _Float16* gptr,
                                              int rows, int strideElems) {
#if TDM_AVAILABLE
  unsigned long long ga = (unsigned long long)(uintptr_t)gptr;
  u32x4 g0;
  g0[0] = 1u;                                          // count=1 (user D#)
  g0[1] = lds_off;                                     // lds_addr (bytes)
  g0[2] = (unsigned)(ga & 0xffffffffu);                // global_addr[31:0]
  g0[3] = (unsigned)((ga >> 32) & 0x01ffffffu) | (2u << 30);  // [56:32]|type=2
  i32x8 g1;
  unsigned K = (unsigned)strideElems, R = (unsigned)rows;
  g1[0] = (int)(1u << 16);                             // data_size=1 -> 2B
  g1[1] = (int)((K & 0xffffu) << 16);                  // tensor_dim0[15:0]
  g1[2] = (int)(((K >> 16) & 0xffffu) | ((R & 0xffffu) << 16)); // dim0 hi|dim1 lo
  g1[3] = (int)(32u << 16);                            // dim1 hi=0 | tile_dim0=32
  g1[4] = (int)(R & 0xffffu);                          // tile_dim1=rows, tile_dim2=0
  g1[5] = (int)K;                                      // tensor_dim0_stride[31:0]
  g1[6] = 0;                                           // stride hi | dim1_stride lo
  g1[7] = 0;
  i32x4 z4 = {0, 0, 0, 0};
#if __clang_major__ >= 23
  i32x8 z8 = {0, 0, 0, 0, 0, 0, 0, 0};
  __builtin_amdgcn_tensor_load_to_lds(g0, g1, z4, z4, z8, 0);
#else
  __builtin_amdgcn_tensor_load_to_lds(g0, g1, z4, z4, 0);
#endif
#else
  (void)lds_off; (void)gptr; (void)rows; (void)strideElems;
#endif
}

// ---------------------------------------------------------------------------
// fp32 -> f16 conversion (weights, once per call; ~120MB -> L2-resident)
// ---------------------------------------------------------------------------
__global__ void cvt_f32_f16(const float* __restrict__ src,
                            _Float16* __restrict__ dst, int n) {
  int i = blockIdx.x * blockDim.x + threadIdx.x;
  if (i < n) dst[i] = (_Float16)src[i];
}

// ---------------------------------------------------------------------------
// Router: rw[t] = hidden[t,:] . router_w
// ---------------------------------------------------------------------------
__global__ void router_kernel(const float* __restrict__ x,
                              const float* __restrict__ w,
                              float* __restrict__ rw) {
  __shared__ float red[256];
  const int t = blockIdx.x;
  const float* row = x + (size_t)t * D_;
  float s = 0.f;
  for (int i = threadIdx.x; i < D_; i += 256) s += row[i] * w[i];
  red[threadIdx.x] = s; __syncthreads();
  for (int o = 128; o > 0; o >>= 1) {
    if (threadIdx.x < o) red[threadIdx.x] += red[threadIdx.x + o];
    __syncthreads();
  }
  if (threadIdx.x == 0) rw[t] = red[0];
}

// ---------------------------------------------------------------------------
// Per-batch k-th largest via 32-pass MSB radix select on order-preserving keys
// ---------------------------------------------------------------------------
__global__ void thresh_kernel(const float* __restrict__ rw,
                              float* __restrict__ thr) {
  __shared__ unsigned int keys[S_];
  __shared__ int cnt;
  const int b = blockIdx.x;
  for (int i = threadIdx.x; i < S_; i += 256) {
    unsigned int u = __float_as_uint(rw[b * S_ + i]);
    keys[i] = (u & 0x80000000u) ? ~u : (u | 0x80000000u);
  }
  __syncthreads();
  unsigned int prefix = 0u;
  for (int bit = 31; bit >= 0; --bit) {
    unsigned int cand = prefix | (1u << bit);
    if (threadIdx.x == 0) cnt = 0;
    __syncthreads();
    int local = 0;
    for (int i = threadIdx.x; i < S_; i += 256) local += (keys[i] >= cand);
    atomicAdd(&cnt, local);
    __syncthreads();
    if (cnt >= KCAP) prefix = cand;
    __syncthreads();
  }
  if (threadIdx.x == 0) {
    unsigned int fb = (prefix & 0x80000000u) ? (prefix & 0x7fffffffu) : ~prefix;
    thr[b] = __uint_as_float(fb);
  }
}

// ---------------------------------------------------------------------------
// RMSNorm (+ optional MoD selection mask) -> f16 activations
// ---------------------------------------------------------------------------
__global__ void rmsnorm_kernel(const float* __restrict__ x,
                               const float* __restrict__ w,
                               _Float16* __restrict__ out,
                               const float* __restrict__ rw,
                               const float* __restrict__ thr,
                               unsigned char* __restrict__ sel,
                               int masked) {
  __shared__ float red[256];
  const int t = blockIdx.x;
  int isSel = 1;
  if (masked) {
    isSel = (rw[t] >= thr[t / S_]) ? 1 : 0;
    if (threadIdx.x == 0) sel[t] = (unsigned char)isSel;
  }
  const float* row = x + (size_t)t * D_;
  float s = 0.f;
  if (isSel)
    for (int i = threadIdx.x; i < D_; i += 256) { float v = row[i]; s += v * v; }
  red[threadIdx.x] = s; __syncthreads();
  for (int o = 128; o > 0; o >>= 1) {
    if (threadIdx.x < o) red[threadIdx.x] += red[threadIdx.x + o];
    __syncthreads();
  }
  const float rs = rsqrtf(red[0] * (1.0f / D_) + 1e-6f);
  for (int i = threadIdx.x; i < D_; i += 256) {
    float v = isSel ? row[i] : 0.f;
    out[(size_t)t * D_ + i] = (_Float16)(v * rs * w[i]);
  }
}

// ---------------------------------------------------------------------------
// Tiled WMMA GEMM: C[M,N] = A[M,K] * W[N,K]^T  (f16 row-major, K-contig).
// Block 256 thr = 8 waves, 128x128 C tile; wave = 64(M) x 32(N) = 8 accums.
// A/W tiles (128x32 f16, 8KB each) are DMA'd into LDS by the Tensor Data
// Mover, double-buffered: issue TDM for tile k+1, compute WMMA on tile k from
// LDS, s_wait_tensorcnt, barrier. Fallback path uses cooperative b128 copies.
// MODE 0: qkv       -> outF = acc + bias(bq|bk|bv)
// MODE 1: o-proj    -> outF = acc + (sel ? hidden : 0)          (residual)
// MODE 2: gate+up   -> outH = silu(accG) * accU    (dual accumulators)
// MODE 3: down+MoD  -> outF = sel ? (resid+acc)*rw : hidden     (final)
// ---------------------------------------------------------------------------
template <int MODE>
__global__ __launch_bounds__(256)
void gemm_kernel(const _Float16* __restrict__ A,
                 const _Float16* __restrict__ W,
                 const _Float16* __restrict__ W2,
                 int M, int N, int K,
                 const float* __restrict__ bq, const float* __restrict__ bk,
                 const float* __restrict__ bv,
                 const float* __restrict__ resid,
                 const float* __restrict__ hidden,
                 const float* __restrict__ rw,
                 const unsigned char* __restrict__ sel,
                 float* __restrict__ outF, _Float16* __restrict__ outH) {
  (void)M; (void)bq; (void)bk; (void)bv; (void)resid; (void)hidden;
  (void)rw; (void)sel; (void)outF; (void)outH; (void)W2;
  const int wave = threadIdx.x >> 5;
  const int lane = threadIdx.x & 31;
  const int mB = blockIdx.y * 128;
  const int nB = blockIdx.x * 128;
  const int mW = (wave & 1) * 64;   // wave offset inside block tile (M)
  const int nW = (wave >> 1) * 32;  // wave offset inside block tile (N)

  __shared__ __align__(16) _Float16 tA[2][128 * 32];
  __shared__ __align__(16) _Float16 tW[2][128 * 32];
  __shared__ __align__(16) _Float16 tW2[(MODE == 2) ? 2 : 1][128 * 32];

  v8f acc[8] = {};
  v8f acc2[8] = {};

  auto issue_tiles = [&](int buf, int k0) {
#if TDM_AVAILABLE
    if (wave == 0) {  // TDM is a per-wave DMA op (EXEC ignored)
      tdm_load_tile((unsigned)(uintptr_t)&tA[buf][0],
                    A + (size_t)mB * (size_t)K + k0, 128, K);
      tdm_load_tile((unsigned)(uintptr_t)&tW[buf][0],
                    W + (size_t)nB * (size_t)K + k0, 128, K);
      if (MODE == 2)
        tdm_load_tile((unsigned)(uintptr_t)&tW2[buf][0],
                      W2 + (size_t)nB * (size_t)K + k0, 128, K);
    }
#else
    for (int i = threadIdx.x; i < 512; i += 256) {
      const int r = i >> 2, c = (i & 3) * 8;
      *(v8h*)&tA[buf][r * 32 + c] =
          *(const v8h*)(A + (size_t)(mB + r) * (size_t)K + k0 + c);
      *(v8h*)&tW[buf][r * 32 + c] =
          *(const v8h*)(W + (size_t)(nB + r) * (size_t)K + k0 + c);
      if (MODE == 2)
        *(v8h*)&tW2[buf][r * 32 + c] =
            *(const v8h*)(W2 + (size_t)(nB + r) * (size_t)K + k0 + c);
    }
#endif
  };
  auto wait_tiles = [&]() {
#if TDM_AVAILABLE
    if (wave == 0) __builtin_amdgcn_s_wait_tensorcnt((short)0);
#endif
    __syncthreads();
  };

  const int nk = K / 32;
  issue_tiles(0, 0);
  wait_tiles();

  for (int kk = 0; kk < nk; ++kk) {
    const int buf = kk & 1;
    if (kk + 1 < nk) issue_tiles(buf ^ 1, (kk + 1) * 32);  // overlap DMA/WMMA

    v16h af[4], bf[2];
#pragma unroll
    for (int mi = 0; mi < 4; ++mi)
      af[mi] = frag_ld(&tA[buf][0], 32, mW + 16 * mi, 0);
#pragma unroll
    for (int ni = 0; ni < 2; ++ni)
      bf[ni] = frag_ld(&tW[buf][0], 32, nW + 16 * ni, 0);
#pragma unroll
    for (int mi = 0; mi < 4; ++mi)
#pragma unroll
      for (int ni = 0; ni < 2; ++ni)
        acc[mi * 2 + ni] = wmma16(af[mi], bf[ni], acc[mi * 2 + ni]);
    if (MODE == 2) {
      v16h bg[2];
#pragma unroll
      for (int ni = 0; ni < 2; ++ni)
        bg[ni] = frag_ld(&tW2[buf][0], 32, nW + 16 * ni, 0);
#pragma unroll
      for (int mi = 0; mi < 4; ++mi)
#pragma unroll
        for (int ni = 0; ni < 2; ++ni)
          acc2[mi * 2 + ni] = wmma16(af[mi], bg[ni], acc2[mi * 2 + ni]);
    }
    wait_tiles();
  }

#pragma unroll
  for (int mi = 0; mi < 4; ++mi)
#pragma unroll
    for (int ni = 0; ni < 2; ++ni)
#pragma unroll
      for (int r = 0; r < 8; ++r) {
        const int row = mB + mW + 16 * mi + r + ((lane & 16) ? 8 : 0);
        const int col = nB + nW + 16 * ni + (lane & 15);
        const size_t idx = (size_t)row * (size_t)N + col;
        const float v = acc[mi * 2 + ni][r];
        if (MODE == 0) {
          float bias = (col < 2048) ? bq[col]
                       : (col < 3072) ? bk[col - 2048] : bv[col - 3072];
          outF[idx] = v + bias;
        } else if (MODE == 1) {
          float res = sel[row] ? hidden[idx] : 0.f;
          outF[idx] = v + res;
        } else if (MODE == 2) {
          float g = v, u = acc2[mi * 2 + ni][r];
          outH[idx] = (_Float16)((g / (1.f + __expf(-g))) * u);
        } else {  // MODE == 3
          float bo = resid[idx] + v;
          outF[idx] = sel[row] ? bo * rw[row] : hidden[idx];
        }
      }
}

// ---------------------------------------------------------------------------
// RoPE + scatter: qkvF[t, 0:2048|2048:3072|3072:4096] -> q16[b,h,s,d],
// k16[b,hk,s,d], vT16[b,hk,d,s] (V transposed so PV B-frags are contiguous)
// ---------------------------------------------------------------------------
__global__ void rope_kernel(const float* __restrict__ qkv,
                            _Float16* __restrict__ q16,
                            _Float16* __restrict__ k16) {
  const int idx = blockIdx.x * blockDim.x + threadIdx.x;
  const int total = NTOK * (HQ_ + HKV_) * 64;
  if (idx >= total) return;
  const int d = idx & 63;
  const int h = (idx >> 6) % (HQ_ + HKV_);
  const int t = idx / (64 * (HQ_ + HKV_));
  const int b = t / S_, s = t % S_;
  const float freq = (float)s * __powf(10000.0f, -(float)(2 * d) * (1.0f / 128.0f));
  float sn, c;
  __sincosf(freq, &sn, &c);
  const float* src;
  _Float16* dst;
  if (h < HQ_) {
    src = qkv + (size_t)t * 4096 + h * HD_;
    dst = q16 + ((size_t)(b * HQ_ + h) * S_ + s) * HD_;
  } else {
    const int hk = h - HQ_;
    src = qkv + (size_t)t * 4096 + 2048 + hk * HD_;
    dst = k16 + ((size_t)(b * HKV_ + hk) * S_ + s) * HD_;
  }
  const float x1 = src[d], x2 = src[d + 64];
  dst[d]      = (_Float16)(x1 * c - x2 * sn);
  dst[d + 64] = (_Float16)(x2 * c + x1 * sn);
}

__global__ void scatv_kernel(const float* __restrict__ qkv,
                             _Float16* __restrict__ vT) {
  const int idx = blockIdx.x * blockDim.x + threadIdx.x;
  if (idx >= NTOK * HKV_ * HD_) return;
  const int d = idx % HD_;
  const int hk = (idx / HD_) % HKV_;
  const int t = idx / (HD_ * HKV_);
  const int b = t / S_, s = t % S_;
  vT[((size_t)(b * HKV_ + hk) * HD_ + d) * S_ + s] =
      (_Float16)qkv[(size_t)t * 4096 + 3072 + hk * HD_ + d];
}

// ---------------------------------------------------------------------------
// Flash attention: 1 wave / 16-query strip, key tiles of 32, online softmax.
// QK^T and P*V both use WMMA; P re-laid out C->A fragments via LDS.
// ---------------------------------------------------------------------------
__global__ __launch_bounds__(32)
void attn_kernel(const _Float16* __restrict__ q16,
                 const _Float16* __restrict__ k16,
                 const _Float16* __restrict__ vT16,
                 _Float16* __restrict__ ctx16) {
  const int q0 = blockIdx.x * 16;
  const int h  = blockIdx.y;
  const int b  = blockIdx.z;
  const int hk = h >> 1;  // HQ/HKV = 2
  const int lane = threadIdx.x;

  const _Float16* Q  = q16  + (size_t)(b * HQ_ + h)  * S_  * HD_;
  const _Float16* Kb = k16  + (size_t)(b * HKV_ + hk) * S_  * HD_;
  const _Float16* Vt = vT16 + (size_t)(b * HKV_ + hk) * HD_ * S_;

  __shared__ __align__(16) _Float16 pS[16 * 32];

  v16h qf[4];
#pragma unroll
  for (int c = 0; c < 4; ++c) qf[c] = frag_ld(Q, HD_, q0, 32 * c);

  v8f o[8] = {};
  float mrow[8], lrow[8];
#pragma unroll
  for (int r = 0; r < 8; ++r) { mrow[r] = -1e30f; lrow[r] = 0.f; }

  const float sc = 0.08838834764831845f;  // 1/sqrt(128)
  const int kend = q0 + 16;               // causal upper bound (exclusive)

  for (int kt = 0; kt < kend; kt += 32) {
    v8f sacc[2] = {};
#pragma unroll
    for (int c = 0; c < 4; ++c) {
      v16h kf0 = frag_ld(Kb, HD_, kt,      32 * c);
      v16h kf1 = frag_ld(Kb, HD_, kt + 16, 32 * c);
      sacc[0] = wmma16(qf[c], kf0, sacc[0]);
      sacc[1] = wmma16(qf[c], kf1, sacc[1]);
    }
    float sv[2][8];
#pragma unroll
    for (int ni = 0; ni < 2; ++ni)
#pragma unroll
      for (int r = 0; r < 8; ++r) {
        const int row = q0 + r + ((lane & 16) ? 8 : 0);
        const int col = kt + 16 * ni + (lane & 15);
        float x = sacc[ni][r] * sc;
        if (col > row) x = -1e9f;
        sv[ni][r] = x;
      }
    float alpha[8];
#pragma unroll
    for (int r = 0; r < 8; ++r) {
      float t = fmaxf(sv[0][r], sv[1][r]);
#pragma unroll
      for (int off = 1; off < 16; off <<= 1) t = fmaxf(t, __shfl_xor(t, off, 32));
      const float mnew = fmaxf(mrow[r], t);
      alpha[r] = __expf(mrow[r] - mnew);
      mrow[r] = mnew;
      float p0 = __expf(sv[0][r] - mnew);
      float p1 = __expf(sv[1][r] - mnew);
      sv[0][r] = p0; sv[1][r] = p1;
      float s = p0 + p1;
#pragma unroll
      for (int off = 1; off < 16; off <<= 1) s += __shfl_xor(s, off, 32);
      lrow[r] = lrow[r] * alpha[r] + s;
    }
#pragma unroll
    for (int nd = 0; nd < 8; ++nd)
#pragma unroll
      for (int r = 0; r < 8; ++r) o[nd][r] *= alpha[r];
#pragma unroll
    for (int ni = 0; ni < 2; ++ni)
#pragma unroll
      for (int r = 0; r < 8; ++r) {
        const int row = r + ((lane & 16) ? 8 : 0);
        const int col = 16 * ni + (lane & 15);
        pS[row * 32 + col] = (_Float16)sv[ni][r];
      }
    __syncthreads();
    v16h pf = frag_ld(pS, 32, 0, 0);
#pragma unroll
    for (int nd = 0; nd < 8; ++nd) {
      v16h vf = frag_ld(Vt, S_, nd * 16, kt);
      o[nd] = wmma16(pf, vf, o[nd]);
    }
    __syncthreads();
  }

#pragma unroll
  for (int nd = 0; nd < 8; ++nd)
#pragma unroll
    for (int r = 0; r < 8; ++r) {
      const int row = q0 + r + ((lane & 16) ? 8 : 0);
      const int col = nd * 16 + (lane & 15);
      const float val = o[nd][r] / lrow[r];
      ctx16[((size_t)(b * S_ + row) * HQ_ + h) * HD_ + col] = (_Float16)val;
    }
}

// ---------------------------------------------------------------------------
// Host-side orchestration
// ---------------------------------------------------------------------------
extern "C" void kernel_launch(void* const* d_in, const int* in_sizes, int n_in,
                              void* d_out, int out_size, void* d_ws, size_t ws_size,
                              hipStream_t stream) {
  (void)in_sizes; (void)n_in; (void)out_size; (void)ws_size;
  const float* hidden   = (const float*)d_in[0];
  const float* router_w = (const float*)d_in[1];
  const float* wq = (const float*)d_in[2];
  const float* bq = (const float*)d_in[3];
  const float* wk = (const float*)d_in[4];
  const float* bk = (const float*)d_in[5];
  const float* wv = (const float*)d_in[6];
  const float* bv = (const float*)d_in[7];
  const float* wo = (const float*)d_in[8];
  const float* wg = (const float*)d_in[9];
  const float* wu = (const float*)d_in[10];
  const float* wd = (const float*)d_in[11];
  const float* ln1 = (const float*)d_in[12];
  const float* ln2 = (const float*)d_in[13];

  char* ws = (char*)d_ws;
  size_t off = 0;
  auto take = [&](size_t bytes) -> void* {
    void* p = ws + off;
    off = (off + bytes + 255) & ~(size_t)255;
    return p;
  };

  float*         rw_d   = (float*)take((size_t)NTOK * 4);
  float*         thr_d  = (float*)take((size_t)B_ * 4);
  unsigned char* sel_d  = (unsigned char*)take((size_t)NTOK);
  _Float16*      h1     = (_Float16*)take((size_t)NTOK * D_ * 2);
  _Float16*      wqkv16 = (_Float16*)take((size_t)4096 * D_ * 2);
  _Float16*      wo16   = (_Float16*)take((size_t)D_ * D_ * 2);
  _Float16*      wg16   = (_Float16*)take((size_t)FF_ * D_ * 2);
  _Float16*      wu16   = (_Float16*)take((size_t)FF_ * D_ * 2);
  _Float16*      wd16   = (_Float16*)take((size_t)D_ * FF_ * 2);
  float*         qkvF   = (float*)take((size_t)NTOK * 4096 * 4);
  _Float16*      q16    = (_Float16*)take((size_t)B_ * HQ_ * S_ * HD_ * 2);
  _Float16*      k16    = (_Float16*)take((size_t)B_ * HKV_ * S_ * HD_ * 2);
  _Float16*      vT16   = (_Float16*)take((size_t)B_ * HKV_ * HD_ * S_ * 2);
  _Float16*      ctx16  = (_Float16*)take((size_t)NTOK * D_ * 2);
  float*         x2     = (float*)take((size_t)NTOK * D_ * 4);
  _Float16*      h2     = (_Float16*)take((size_t)NTOK * D_ * 2);
  _Float16*      a16    = (_Float16*)take((size_t)NTOK * FF_ * 2);

  auto cvt = [&](const float* s, _Float16* d, int n) {
    cvt_f32_f16<<<(n + 255) / 256, 256, 0, stream>>>(s, d, n);
  };
  // Weight conversions (wq|wk|wv packed contiguously so QKV is one GEMM)
  cvt(wq, wqkv16,               2048 * 2048);
  cvt(wk, wqkv16 + 2048 * 2048, 1024 * 2048);
  cvt(wv, wqkv16 + 3072 * 2048, 1024 * 2048);
  cvt(wo, wo16, 2048 * 2048);
  cvt(wg, wg16, FF_ * 2048);
  cvt(wu, wu16, FF_ * 2048);
  cvt(wd, wd16, 2048 * FF_);

  // Router + MoD threshold
  router_kernel<<<NTOK, 256, 0, stream>>>(hidden, router_w, rw_d);
  thresh_kernel<<<B_, 256, 0, stream>>>(rw_d, thr_d);

  // RMSNorm-1 with selection mask
  rmsnorm_kernel<<<NTOK, 256, 0, stream>>>(hidden, ln1, h1, rw_d, thr_d, sel_d, 1);

  // QKV GEMM: [4096,2048] x [4096,2048]^T -> qkvF [4096,4096]
  gemm_kernel<0><<<dim3(4096 / 128, NTOK / 128), 256, 0, stream>>>(
      h1, wqkv16, nullptr, NTOK, 4096, 2048,
      bq, bk, bv, nullptr, nullptr, nullptr, nullptr, qkvF, nullptr);

  // RoPE + V transpose/scatter
  {
    int n = NTOK * (HQ_ + HKV_) * 64;
    rope_kernel<<<(n + 255) / 256, 256, 0, stream>>>(qkvF, q16, k16);
    int nv = NTOK * HKV_ * HD_;
    scatv_kernel<<<(nv + 255) / 256, 256, 0, stream>>>(qkvF, vT16);
  }

  // Flash attention (causal, GQA 2:1)
  attn_kernel<<<dim3(S_ / 16, HQ_, B_), 32, 0, stream>>>(q16, k16, vT16, ctx16);

  // O-proj + residual (sel_hidden recomputed from mask in epilogue)
  gemm_kernel<1><<<dim3(2048 / 128, NTOK / 128), 256, 0, stream>>>(
      ctx16, wo16, nullptr, NTOK, 2048, 2048,
      nullptr, nullptr, nullptr, nullptr, hidden, nullptr, sel_d, x2, nullptr);

  // RMSNorm-2
  rmsnorm_kernel<<<NTOK, 256, 0, stream>>>(x2, ln2, h2, nullptr, nullptr, nullptr, 0);

  // Fused gate+up GEMM -> a16 = silu(g)*u (reads A frags once)
  gemm_kernel<2><<<dim3(FF_ / 128, NTOK / 128), 256, 0, stream>>>(
      h2, wg16, wu16, NTOK, FF_, 2048,
      nullptr, nullptr, nullptr, nullptr, nullptr, nullptr, nullptr, nullptr, a16);

  // Down GEMM + residual + MoD combine -> d_out
  gemm_kernel<3><<<dim3(2048 / 128, NTOK / 128), 256, 0, stream>>>(
      a16, wd16, nullptr, NTOK, 2048, FF_,
      nullptr, nullptr, nullptr, x2, hidden, rw_d, sel_d, (float*)d_out, nullptr);
}
